// SG_Self_Attention_24026047054362
// MI455X (gfx1250) — matrix operand order repaired and checked by reading, not compiled, not run on hardware
//
#include <hip/hip_runtime.h>
#include <hip/hip_bf16.h>
#include <math.h>

typedef __attribute__((ext_vector_type(16))) _Float16 v16h;
typedef __attribute__((ext_vector_type(8)))  _Float16 v8h;
typedef __attribute__((ext_vector_type(8)))  float    v8f;

#define BB 4
#define HH 64
#define WW 64
#define CC 256
#define NN 4096      // HH*WW
#define H2 4         // NH/2
#define DD 32        // head dim
#define N1 1024      // (H/2)*(W/2)
#define OUT_ELEMS (BB*NN*CC)

union V16U { v16h v; v8h h[2]; };
union U4H  { uint4 u; _Float16 h[8]; };

// ---------------------------------------------------------------------------
// helpers
// ---------------------------------------------------------------------------
__device__ inline float redmax16(float v) {
    v = fmaxf(v, __shfl_xor(v, 1, 32));
    v = fmaxf(v, __shfl_xor(v, 2, 32));
    v = fmaxf(v, __shfl_xor(v, 4, 32));
    v = fmaxf(v, __shfl_xor(v, 8, 32));
    return v;
}
__device__ inline float redsum16(float v) {
    v += __shfl_xor(v, 1, 32);
    v += __shfl_xor(v, 2, 32);
    v += __shfl_xor(v, 4, 32);
    v += __shfl_xor(v, 8, 32);
    return v;
}

// Convert 16 contiguous f32 -> v16h fragment lane data (4x float4 loads).
__device__ inline v16h cvt16_f32(const float* __restrict__ p) {
    v16h r;
    const float4* q = (const float4*)p;
#pragma unroll
    for (int c = 0; c < 4; ++c) {
        float4 f = q[c];
        r[c * 4 + 0] = (_Float16)f.x;
        r[c * 4 + 1] = (_Float16)f.y;
        r[c * 4 + 2] = (_Float16)f.z;
        r[c * 4 + 3] = (_Float16)f.w;
    }
    return r;
}

// A-layout (16x32 f16): lane l holds row M=l%16; halves j<8 -> K=8*(l/16)+j,
// halves j>=8 -> K=16+8*(l/16)+(j-8).  P tile is 16x16 (K>=16 zero padded).
__device__ inline v16h load_A_p16(const _Float16* s /*16x16 row-major*/) {
    int l = threadIdx.x & 31;
    int m = l & 15;
    int kb = (l >> 4) * 8;
    V16U a;
    a.h[0] = *(const v8h*)&s[m * 16 + kb];      // 16B aligned
#pragma unroll
    for (int j = 0; j < 8; ++j) a.v[8 + j] = (_Float16)0.0f;  // K >= 16
    return a.v;
}

// ---------------------------------------------------------------------------
// generic WMMA GEMM:  Y[M,N] = alpha * (X[M,K](f16) @ W[K,N](f16) + bias[N])
// block = 128 threads (4 waves), grid = (M/64, N/64)
// LDS: X tile row-major (64x32), W tile TRANSPOSED (WsT[n][k], stride 48) so
// both A and B fragments load as ds_load_b128.
// ---------------------------------------------------------------------------
__global__ void gemm_f16_kernel(const _Float16* __restrict__ X,
                                const _Float16* __restrict__ W,
                                const float* __restrict__ bias,
                                float* __restrict__ Y,
                                int M, int N, int K, float alpha) {
    __shared__ _Float16 Xs[64 * 32];
    __shared__ _Float16 WsT[64 * 48];   // [n][k], padded stride 48 halves
    const int tid = threadIdx.x;
    const int l   = tid & 31;
    const int wv  = tid >> 5;
    const int m0  = blockIdx.x * 64;
    const int n0  = blockIdx.y * 64;

    v8f acc[4] = {};

    for (int k0 = 0; k0 < K; k0 += 32) {
        __syncthreads();
        // X tile: 64 rows x 32 halves, 16B vector copies (coalesced both ways)
        for (int c = tid; c < 256; c += 128) {
            int r = c >> 2, kc = (c & 3) * 8;
            *(uint4*)&Xs[r * 32 + kc] =
                *(const uint4*)&X[(size_t)(m0 + r) * K + k0 + kc];
        }
        // W tile: global b128 loads (coalesced over n), transposed scatter to LDS
        for (int c = tid; c < 256; c += 128) {
            int k = c >> 3, nc = (c & 7) * 8;
            U4H t;
            t.u = *(const uint4*)&W[(size_t)(k0 + k) * N + n0 + nc];
#pragma unroll
            for (int j = 0; j < 8; ++j) WsT[(nc + j) * 48 + k] = t.h[j];
        }
        if (k0 + 32 < K) {
            __builtin_prefetch(&X[(size_t)(m0 + (tid >> 1)) * K + k0 + 32], 0, 0);
            __builtin_prefetch(&W[(size_t)(k0 + 32 + (tid >> 2)) * N + n0], 0, 0);
        }
        __syncthreads();

        // A fragment: two contiguous 16B LDS reads
        V16U a;
        {
            int m = wv * 16 + (l & 15);
            int kb = (l >> 4) * 8;
            a.h[0] = *(const v8h*)&Xs[m * 32 + kb];
            a.h[1] = *(const v8h*)&Xs[m * 32 + 16 + kb];
        }
#pragma unroll
        for (int t = 0; t < 4; ++t) {
            // B fragment: two contiguous 16B LDS reads from transposed tile
            V16U bm;
            int n  = t * 16 + (l & 15);
            int kb = (l >> 4) * 8;
            bm.h[0] = *(const v8h*)&WsT[n * 48 + kb];
            bm.h[1] = *(const v8h*)&WsT[n * 48 + 16 + kb];
            acc[t] = __builtin_amdgcn_wmma_f32_16x16x32_f16(
                false, a.v, false, bm.v, (short)0, acc[t], false, false);
        }
    }

#pragma unroll
    for (int t = 0; t < 4; ++t) {
#pragma unroll
        for (int r = 0; r < 8; ++r) {
            int row = m0 + wv * 16 + r + ((l >> 4) << 3);
            int col = n0 + t * 16 + (l & 15);
            Y[(size_t)row * N + col] = alpha * (acc[t][r] + bias[col]);
        }
    }
}

// ---------------------------------------------------------------------------
// attention branch 1: SR global attention.  One wave per 16-query tile per
// (b, head).  grid = 1024 blocks * 128 threads (4 independent waves).
// ---------------------------------------------------------------------------
__device__ inline v8f qk_tile(const v16h& aq, const float* __restrict__ kbase,
                              int key0) {
    int l = threadIdx.x & 31;
    int key = key0 + (l & 15);
    int db  = (l >> 4) * 16;
    v16h bk = cvt16_f32(&kbase[(size_t)key * CC + db]);
    v8f s = {};
    return __builtin_amdgcn_wmma_f32_16x16x32_f16(false, aq, false, bk,
                                                  (short)0, s, false, false);
}

__global__ void attn1_kernel(const float* __restrict__ Q1,
                             const float* __restrict__ KV1,
                             float* __restrict__ cc,
                             float* __restrict__ g_acc) {
    __shared__ _Float16 Pt[4][16 * 16];
    const int tid = threadIdx.x, l = tid & 31, wv = tid >> 5;
    const int wid = blockIdx.x * 4 + wv;
    const int b  = wid >> 10;
    const int h  = (wid >> 8) & 3;
    const int qt = wid & 255;
    const float scale = 0.17677669529663687f;  // 1/sqrt(32)

    const float* qbase = Q1 + ((size_t)(b * NN + qt * 16)) * (CC / 2) + h * DD;
    const float* kbase = KV1 + (size_t)b * N1 * CC + h * DD;
    const float* vbase = KV1 + (size_t)b * N1 * CC + 128 + h * DD;

    // A regs: Q tile * scale (contiguous f32 runs -> float4 loads)
    v16h aq;
    {
        int m = l & 15, kb = (l >> 4) * 16;
        const float* qr = qbase + (size_t)m * (CC / 2);
        float4 f0 = *(const float4*)&qr[kb];
        float4 f1 = *(const float4*)&qr[kb + 4];
        float4 f2 = *(const float4*)&qr[kb + 8];
        float4 f3 = *(const float4*)&qr[kb + 12];
        aq[0] = (_Float16)(f0.x * scale); aq[1] = (_Float16)(f0.y * scale);
        aq[2] = (_Float16)(f0.z * scale); aq[3] = (_Float16)(f0.w * scale);
        aq[4] = (_Float16)(f1.x * scale); aq[5] = (_Float16)(f1.y * scale);
        aq[6] = (_Float16)(f1.z * scale); aq[7] = (_Float16)(f1.w * scale);
        aq[8]  = (_Float16)(f2.x * scale); aq[9]  = (_Float16)(f2.y * scale);
        aq[10] = (_Float16)(f2.z * scale); aq[11] = (_Float16)(f2.w * scale);
        aq[12] = (_Float16)(f3.x * scale); aq[13] = (_Float16)(f3.y * scale);
        aq[14] = (_Float16)(f3.z * scale); aq[15] = (_Float16)(f3.w * scale);
    }

    // pass 1: row max
    float mrow[8];
#pragma unroll
    for (int r = 0; r < 8; ++r) mrow[r] = -1e30f;
    for (int c = 0; c < 64; ++c) {
        v8f s = qk_tile(aq, kbase, c * 16);
#pragma unroll
        for (int r = 0; r < 8; ++r) mrow[r] = fmaxf(mrow[r], s[r]);
    }
#pragma unroll
    for (int r = 0; r < 8; ++r) mrow[r] = redmax16(mrow[r]);

    // pass 2: exp-sums + O = P @ V (unnormalized)
    float lrow[8];
#pragma unroll
    for (int r = 0; r < 8; ++r) lrow[r] = 0.0f;
    v8f o0 = {}, o1 = {};
    for (int c = 0; c < 64; ++c) {
        v8f s = qk_tile(aq, kbase, c * 16);
        float p[8];
#pragma unroll
        for (int r = 0; r < 8; ++r) {
            p[r] = __expf(s[r] - mrow[r]);
            lrow[r] += p[r];
        }
        __syncthreads();
#pragma unroll
        for (int r = 0; r < 8; ++r)
            Pt[wv][(r + ((l >> 4) << 3)) * 16 + (l & 15)] = (_Float16)p[r];
        __syncthreads();
        v16h ap = load_A_p16(&Pt[wv][0]);
        v16h bv0, bv1;
        {
            int grp = l >> 4, dd = l & 15;
#pragma unroll
            for (int j = 0; j < 16; ++j) {
                float f0 = 0.0f, f1 = 0.0f;
                if (j < 8) {
                    const float* vr = vbase + (size_t)(c * 16 + grp * 8 + j) * CC;
                    f0 = vr[dd];
                    f1 = vr[16 + dd];
                }
                bv0[j] = (_Float16)f0;
                bv1[j] = (_Float16)f1;
            }
        }
        o0 = __builtin_amdgcn_wmma_f32_16x16x32_f16(false, ap, false, bv0,
                                                    (short)0, o0, false, false);
        o1 = __builtin_amdgcn_wmma_f32_16x16x32_f16(false, ap, false, bv1,
                                                    (short)0, o1, false, false);
    }
#pragma unroll
    for (int r = 0; r < 8; ++r) lrow[r] = redsum16(lrow[r]);

    // store normalized O into concat buffer (channels [0,128))
#pragma unroll
    for (int r = 0; r < 8; ++r) {
        float inv = 1.0f / lrow[r];
        int row   = qt * 16 + r + ((l >> 4) << 3);
        float* cr = cc + ((size_t)(b * NN + row)) * CC + h * DD;
        cr[l & 15]        = o0[r] * inv;
        cr[16 + (l & 15)] = o1[r] * inv;
    }

    // pass 3: column sums of normalized attention -> g
    for (int c = 0; c < 64; ++c) {
        v8f s = qk_tile(aq, kbase, c * 16);
        float cs = 0.0f;
#pragma unroll
        for (int r = 0; r < 8; ++r) cs += __expf(s[r] - mrow[r]) / lrow[r];
        atomicAdd(&g_acc[b * N1 + c * 16 + (l & 15)], cs);
    }
}

// ---------------------------------------------------------------------------
// attention branch 2: 4x4 windows, 16 tokens, d=32.  One wave per
// (b, head, window); grid = 1024 blocks * 128 threads.
// ---------------------------------------------------------------------------
__global__ void attn2_kernel(const float* __restrict__ Q2,
                             const float* __restrict__ KV2,
                             float* __restrict__ cc,
                             float* __restrict__ lm_acc) {
    __shared__ _Float16 Pt[4][16 * 16];
    const int tid = threadIdx.x, l = tid & 31, wv = tid >> 5;
    const int wid = blockIdx.x * 4 + wv;
    const int b   = wid >> 10;
    const int h   = (wid >> 8) & 3;
    const int win = wid & 255;
    const int wh = win >> 4, wwi = win & 15;
    const float scale = 0.17677669529663687f;

    auto tok_n = [&](int t) {
        int r = t >> 2, cx = t & 3;
        return (wh * 4 + r) * WW + wwi * 4 + cx;
    };

    v16h aq;
    {
        int m = l & 15, kb = (l >> 4) * 16;
        int n = tok_n(m);
        const float* qr = Q2 + ((size_t)(b * NN + n)) * (CC / 2) + h * DD;
#pragma unroll
        for (int j = 0; j < 8; ++j) aq[j] = (_Float16)(qr[kb + j] * scale);
#pragma unroll
        for (int j = 0; j < 8; ++j) aq[8 + j] = (_Float16)(qr[kb + 8 + j] * scale);
    }
    v16h bk;
    {
        int key = l & 15, db = (l >> 4) * 16;
        int n = tok_n(key);
        bk = cvt16_f32(KV2 + ((size_t)(b * NN + n)) * CC + h * DD + db);
    }
    v8f s = {};
    s = __builtin_amdgcn_wmma_f32_16x16x32_f16(false, aq, false, bk, (short)0,
                                               s, false, false);
    // softmax over 16 columns per row
    float p[8];
#pragma unroll
    for (int r = 0; r < 8; ++r) {
        float mr = redmax16(s[r]);
        p[r] = __expf(s[r] - mr);
        float sm = redsum16(p[r]);
        p[r] /= sm;
    }
    // lm accumulation (mean over heads + rows handled at assembly)
    float cs = 0.0f;
#pragma unroll
    for (int r = 0; r < 8; ++r) cs += p[r];
    atomicAdd(&lm_acc[(b * 256 + win) * 16 + (l & 15)], cs);

    __syncthreads();
#pragma unroll
    for (int r = 0; r < 8; ++r)
        Pt[wv][(r + ((l >> 4) << 3)) * 16 + (l & 15)] = (_Float16)p[r];
    __syncthreads();
    v16h ap = load_A_p16(&Pt[wv][0]);
    v16h bv0, bv1;
    {
        int grp = l >> 4, dd = l & 15;
#pragma unroll
        for (int j = 0; j < 16; ++j) {
            float f0 = 0.0f, f1 = 0.0f;
            if (j < 8) {
                int n = tok_n(grp * 8 + j);
                const float* vr = KV2 + ((size_t)(b * NN + n)) * CC + 128 + h * DD;
                f0 = vr[dd];
                f1 = vr[16 + dd];
            }
            bv0[j] = (_Float16)f0;
            bv1[j] = (_Float16)f1;
        }
    }
    v8f o0 = {}, o1 = {};
    o0 = __builtin_amdgcn_wmma_f32_16x16x32_f16(false, ap, false, bv0, (short)0,
                                                o0, false, false);
    o1 = __builtin_amdgcn_wmma_f32_16x16x32_f16(false, ap, false, bv1, (short)0,
                                                o1, false, false);
#pragma unroll
    for (int r = 0; r < 8; ++r) {
        int t = r + ((l >> 4) << 3);
        int n = tok_n(t);
        float* cr = cc + ((size_t)(b * NN + n)) * CC + 128 + h * DD;
        cr[l & 15]        = o0[r];
        cr[16 + (l & 15)] = o1[r];
    }
}

// ---------------------------------------------------------------------------
// elementwise helpers
// ---------------------------------------------------------------------------
__global__ void cvt_f32_f16(const float* __restrict__ in,
                            _Float16* __restrict__ out, int n) {
    int i = blockIdx.x * 256 + threadIdx.x;
    if (i < n) out[i] = (_Float16)in[i];
}

__global__ void zero_f32(float* __restrict__ p, int n) {
    int i = blockIdx.x * 256 + threadIdx.x;
    if (i < n) p[i] = 0.0f;
}

__global__ void srw_transpose(const float* __restrict__ srw,
                              _Float16* __restrict__ out) {
    int i = blockIdx.x * 256 + threadIdx.x;  // over 256*1024
    if (i >= CC * 1024) return;
    int co = i >> 10, k = i & 1023;
    out[k * CC + co] = (_Float16)srw[i];
}

__global__ void patch_kernel(const float* __restrict__ x,
                             _Float16* __restrict__ xp) {
    int i = blockIdx.x * 256 + threadIdx.x;  // over B*1024*1024
    if (i >= BB * N1 * 1024) return;
    int k  = i & 1023;
    int n1 = (i >> 10) & 1023;
    int b  = i >> 20;
    int ci = k >> 2, kh = (k >> 1) & 1, kw = k & 1;
    int oh = n1 >> 5, ow = n1 & 31;
    int n  = (2 * oh + kh) * WW + (2 * ow + kw);
    xp[i]  = (_Float16)x[((size_t)(b * NN + n)) * CC + ci];
}

__global__ void ln_gelu_kernel(const float* __restrict__ xin,
                               const float* __restrict__ w,
                               const float* __restrict__ bb,
                               _Float16* __restrict__ out) {
    __shared__ float red[256];
    int t = blockIdx.x;     // token (B*N1)
    int c = threadIdx.x;
    float v = xin[(size_t)t * CC + c];
    red[c] = v;
    __syncthreads();
    for (int s = 128; s > 0; s >>= 1) {
        if (c < s) red[c] += red[c + s];
        __syncthreads();
    }
    float mean = red[0] / 256.0f;
    __syncthreads();
    float d = v - mean;
    red[c] = d * d;
    __syncthreads();
    for (int s = 128; s > 0; s >>= 1) {
        if (c < s) red[c] += red[c + s];
        __syncthreads();
    }
    float var = red[0] / 256.0f;
    float y = d * rsqrtf(var + 1e-5f) * w[c] + bb[c];
    float g = 0.5f * y * (1.0f + erff(y * 0.70710678118654752f));
    out[(size_t)t * CC + c] = (_Float16)g;
}

__global__ void dwconv_kernel(const float* __restrict__ Yl,
                              const float* __restrict__ wc,
                              const float* __restrict__ bc,
                              float* __restrict__ lepe) {
    int idx = blockIdx.x * 256 + threadIdx.x;  // over B*N*C
    if (idx >= OUT_ELEMS) return;
    int c  = idx & 255;
    int n  = (idx >> 8) & 4095;
    int b  = idx >> 20;
    int hh = n >> 6, ww2 = n & 63;
    float acc = bc[c];
#pragma unroll
    for (int dh = -1; dh <= 1; ++dh)
#pragma unroll
        for (int dw = -1; dw <= 1; ++dw) {
            int y = hh + dh, x = ww2 + dw;
            if (y >= 0 && y < HH && x >= 0 && x < WW)
                acc += Yl[((size_t)(b * NN + y * WW + x)) * CC + c] *
                       wc[c * 9 + (dh + 1) * 3 + (dw + 1)];
        }
    lepe[idx] = acc;
}

__global__ void addcvt_kernel(const float* __restrict__ cc,
                              const float* __restrict__ lepe,
                              _Float16* __restrict__ out, int n) {
    int i = blockIdx.x * 256 + threadIdx.x;
    if (i < n) out[i] = (_Float16)(cc[i] + lepe[i]);
}

__global__ void mask_kernel(const float* __restrict__ g_acc,
                            const float* __restrict__ lm_acc,
                            float* __restrict__ out) {
    int idx = blockIdx.x * 256 + threadIdx.x;  // over B*H*W
    if (idx >= BB * NN) return;
    int b   = idx >> 12;
    int pix = idx & 4095;
    int hh = pix >> 6, ww = pix & 63;
    int win = (hh >> 2) * 16 + (ww >> 2);
    int col = (hh & 3) * 4 + (ww & 3);
    float lmv = lm_acc[(b * 256 + win) * 16 + col] * (1.0f / 64.0f);
    float gv  = g_acc[b * N1 + (hh >> 1) * 32 + (ww >> 1)] * (1.0f / 16384.0f);
    float m = lmv + gv;
    out[OUT_ELEMS + idx] = m;                                   // mask1
    out[OUT_ELEMS + BB * NN + (b << 12) + ww * 64 + hh] = m;    // mask2
}

// ---------------------------------------------------------------------------
// launch
// ---------------------------------------------------------------------------
extern "C" void kernel_launch(void* const* d_in, const int* in_sizes, int n_in,
                              void* d_out, int out_size, void* d_ws,
                              size_t ws_size, hipStream_t stream) {
    const float* x       = (const float*)d_in[0];
    const float* q1_w    = (const float*)d_in[1];
    const float* q1_b    = (const float*)d_in[2];
    const float* kv1_w   = (const float*)d_in[3];
    const float* kv1_b   = (const float*)d_in[4];
    const float* q2_w    = (const float*)d_in[5];
    const float* q2_b    = (const float*)d_in[6];
    const float* kv2_w   = (const float*)d_in[7];
    const float* kv2_b   = (const float*)d_in[8];
    const float* lepe_w  = (const float*)d_in[9];
    const float* lepe_b  = (const float*)d_in[10];
    const float* lconv_w = (const float*)d_in[11];
    const float* lconv_b = (const float*)d_in[12];
    const float* sr_w    = (const float*)d_in[13];
    const float* sr_b    = (const float*)d_in[14];
    const float* norm_w  = (const float*)d_in[15];
    const float* norm_b  = (const float*)d_in[16];
    const float* proj_w  = (const float*)d_in[17];
    const float* proj_b  = (const float*)d_in[18];
    float* out = (float*)d_out;

    char* base = (char*)d_ws;
    size_t off = 0;
    auto alloc = [&](size_t bytes) -> void* {
        void* p = base + off;
        off += (bytes + 255) & ~(size_t)255;
        return p;
    };

    _Float16* xh      = (_Float16*)alloc((size_t)OUT_ELEMS * 2);
    _Float16* wlepe_h = (_Float16*)alloc(CC * CC * 2);
    _Float16* wq1_h   = (_Float16*)alloc(CC * 128 * 2);
    _Float16* wq2_h   = (_Float16*)alloc(CC * 128 * 2);
    _Float16* wkv1_h  = (_Float16*)alloc(CC * CC * 2);
    _Float16* wkv2_h  = (_Float16*)alloc(CC * CC * 2);
    _Float16* wproj_h = (_Float16*)alloc(CC * CC * 2);
    _Float16* wsr_h   = (_Float16*)alloc(1024 * CC * 2);
    float*    Yl      = (float*)alloc((size_t)OUT_ELEMS * 4);
    float*    lepe    = (float*)alloc((size_t)OUT_ELEMS * 4);
    float*    Q1      = (float*)alloc((size_t)BB * NN * 128 * 4);
    float*    Q2      = (float*)alloc((size_t)BB * NN * 128 * 4);
    float*    KV2     = (float*)alloc((size_t)OUT_ELEMS * 4);
    _Float16* Xp      = (_Float16*)alloc((size_t)BB * N1 * 1024 * 2);
    float*    XSpre   = (float*)alloc((size_t)BB * N1 * CC * 4);
    _Float16* xs_h    = (_Float16*)alloc((size_t)BB * N1 * CC * 2);
    float*    KV1     = (float*)alloc((size_t)BB * N1 * CC * 4);
    float*    cc      = (float*)alloc((size_t)OUT_ELEMS * 4);
    _Float16* cc_h    = (_Float16*)alloc((size_t)OUT_ELEMS * 2);
    float*    g_acc   = (float*)alloc(BB * N1 * 4);
    float*    lm_acc  = (float*)alloc(BB * 256 * 16 * 4);

    auto blk = [](int n) { return (n + 255) / 256; };

    // conversions + zeroing
    cvt_f32_f16<<<blk(OUT_ELEMS), 256, 0, stream>>>(x, xh, OUT_ELEMS);
    cvt_f32_f16<<<blk(CC * CC), 256, 0, stream>>>(lepe_w, wlepe_h, CC * CC);
    cvt_f32_f16<<<blk(CC * 128), 256, 0, stream>>>(q1_w, wq1_h, CC * 128);
    cvt_f32_f16<<<blk(CC * 128), 256, 0, stream>>>(q2_w, wq2_h, CC * 128);
    cvt_f32_f16<<<blk(CC * CC), 256, 0, stream>>>(kv1_w, wkv1_h, CC * CC);
    cvt_f32_f16<<<blk(CC * CC), 256, 0, stream>>>(kv2_w, wkv2_h, CC * CC);
    cvt_f32_f16<<<blk(CC * CC), 256, 0, stream>>>(proj_w, wproj_h, CC * CC);
    srw_transpose<<<blk(CC * 1024), 256, 0, stream>>>(sr_w, wsr_h);
    zero_f32<<<blk(BB * N1), 256, 0, stream>>>(g_acc, BB * N1);
    zero_f32<<<blk(BB * 256 * 16), 256, 0, stream>>>(lm_acc, BB * 256 * 16);

    const int M = BB * NN;  // 16384
    // lepe linear, q1, q2, kv2
    gemm_f16_kernel<<<dim3(M / 64, CC / 64), 128, 0, stream>>>(
        xh, wlepe_h, lepe_b, Yl, M, CC, CC, 1.0f);
    gemm_f16_kernel<<<dim3(M / 64, 128 / 64), 128, 0, stream>>>(
        xh, wq1_h, q1_b, Q1, M, 128, CC, 1.0f);
    gemm_f16_kernel<<<dim3(M / 64, 128 / 64), 128, 0, stream>>>(
        xh, wq2_h, q2_b, Q2, M, 128, CC, 1.0f);
    gemm_f16_kernel<<<dim3(M / 64, CC / 64), 128, 0, stream>>>(
        xh, wkv2_h, kv2_b, KV2, M, CC, CC, 1.0f);

    // depthwise conv on lepe
    dwconv_kernel<<<blk(OUT_ELEMS), 256, 0, stream>>>(Yl, lconv_w, lconv_b, lepe);

    // SR conv as im2col GEMM -> LN+GELU -> kv1
    patch_kernel<<<blk(BB * N1 * 1024), 256, 0, stream>>>(x, Xp);
    gemm_f16_kernel<<<dim3((BB * N1) / 64, CC / 64), 128, 0, stream>>>(
        Xp, wsr_h, sr_b, XSpre, BB * N1, CC, 1024, 1.0f);
    ln_gelu_kernel<<<BB * N1, 256, 0, stream>>>(XSpre, norm_w, norm_b, xs_h);
    gemm_f16_kernel<<<dim3((BB * N1) / 64, CC / 64), 128, 0, stream>>>(
        xs_h, wkv1_h, kv1_b, KV1, BB * N1, CC, CC, 1.0f);

    // attention branches
    attn1_kernel<<<1024, 128, 0, stream>>>(Q1, KV1, cc, g_acc);
    attn2_kernel<<<1024, 128, 0, stream>>>(Q2, KV2, cc, lm_acc);

    // (concat + lepe) -> f16 -> projection with x2 scale
    addcvt_kernel<<<blk(OUT_ELEMS), 256, 0, stream>>>(cc, lepe, cc_h, OUT_ELEMS);
    gemm_f16_kernel<<<dim3(M / 64, CC / 64), 128, 0, stream>>>(
        cc_h, wproj_h, proj_b, out, M, CC, CC, 2.0f);

    // masks
    mask_kernel<<<blk(BB * NN), 256, 0, stream>>>(g_acc, lm_acc, out);
}
